// MultiheadAttentionWithRelativeBias_65077344468995
// MI455X (gfx1250) — matrix operand order repaired and checked by reading, not compiled, not run on hardware
//
#include <hip/hip_runtime.h>

#define S_LEN 2048
#define BATCH 4
#define EMB   512
#define NH    8
#define HDIM  64
#define E3    1536
#define SC_LD 2052   // 2048 + 4 floats pad: row stride = 4 banks -> conflict-free row-parallel reads

typedef __attribute__((ext_vector_type(16))) _Float16 v16h;
typedef __attribute__((ext_vector_type(8)))  _Float16 v8h;
typedef __attribute__((ext_vector_type(8)))  float    v8f;
typedef __attribute__((ext_vector_type(4)))  float    v4f;

// ---------------- WMMA fragment loaders (wave32, 16x16x32 f16) ----------------
// A (16x32 f16): lane L holds row (L&15); halves 0..7 = K {0..7}+8*(L>=16),
// halves 8..15 = K {16..23}+8*(L>=16).
static __device__ __forceinline__ v16h load_a16(const _Float16* __restrict__ base,
                                                int row0, int ld, int k0, int lane) {
  const _Float16* p = base + (size_t)(row0 + (lane & 15)) * (size_t)ld
                      + k0 + (((lane >> 4) & 1) << 3);
  union { v16h v; v8h h[2]; } u;
  u.h[0] = *(const v8h*)(p);
  u.h[1] = *(const v8h*)(p + 16);
  return u.v;
}

// B (32x16 f16) where B[k][n] = W[n0+n][k] (row-major W, i.e. B = W^T):
// lane n holds column (n&15), halves 0..7 = K {0..7}+8*(n>=16), halves 8..15 = K {16..23}+8*(n>=16).
static __device__ __forceinline__ v16h load_bt16(const _Float16* __restrict__ w,
                                                 int n0, int ld, int k0, int lane) {
  const _Float16* p = w + (size_t)(n0 + (lane & 15)) * (size_t)ld
                      + k0 + (((lane >> 4) & 1) << 3);
  union { v16h v; v8h h[2]; } u;
  u.h[0] = *(const v8h*)(p);
  u.h[1] = *(const v8h*)(p + 16);
  return u.v;
}

static __device__ __forceinline__ v8f wmma_f16(v16h a, v16h b, v8f c) {
  return __builtin_amdgcn_wmma_f32_16x16x32_f16(false, a, false, b, (short)0, c,
                                                false, false);
}

// ---------------- K0: fp32 -> fp16 convert ----------------
__global__ void k_cvt_f16(const float* __restrict__ src, _Float16* __restrict__ dst, int n) {
  int i = blockIdx.x * blockDim.x + threadIdx.x;
  if (i < n) dst[i] = (_Float16)src[i];
}

// ---------------- K1: QKV projection GEMM (8192 x 1536 x 512), 32x32 per wave ----------------
// rows m = s*BATCH + b (query is (S,B,E)).  q,k -> head-major [bh][s][hd]; v -> transposed [bh][hd][s].
static __device__ __forceinline__ void store_qkv_tile(const v8f& acc, int m0, int n0, int lane,
                                                      const float* __restrict__ bin,
                                                      _Float16* __restrict__ qh,
                                                      _Float16* __restrict__ kh,
                                                      _Float16* __restrict__ vt) {
  const int n = n0 + (lane & 15);
  const float bias = bin[n];
  const int which = n >> 9;          // 0=q 1=k 2=v
  const int rem = n & 511;
  const int h = rem >> 6, hd = rem & 63;
  const int hi8 = ((lane >> 4) & 1) << 3;
#pragma unroll
  for (int r = 0; r < 8; ++r) {
    const int m = m0 + r + hi8;
    const int s = m >> 2;            // m = s*4 + b
    const int bb = m & 3;
    const _Float16 val = (_Float16)(acc[r] + bias);
    if (which == 2) {
      vt[(((size_t)bb * NH + h) * HDIM + hd) * S_LEN + s] = val;
    } else {
      _Float16* dst = (which == 0) ? qh : kh;
      dst[((((size_t)bb * NH + h) * S_LEN) + s) * HDIM + hd] = val;
    }
  }
}

__global__ void k_qkv_gemm(const _Float16* __restrict__ qin, const _Float16* __restrict__ win,
                           const float* __restrict__ bin,
                           _Float16* __restrict__ qh, _Float16* __restrict__ kh,
                           _Float16* __restrict__ vt) {
  const int lane = threadIdx.x;
  const int m0 = blockIdx.x * 32;
  const int n0 = blockIdx.y * 32;
  v8f acc00 = {}, acc01 = {}, acc10 = {}, acc11 = {};
#pragma unroll 2
  for (int k0 = 0; k0 < EMB; k0 += 32) {
    v16h a0 = load_a16(qin, m0,      EMB, k0, lane);
    v16h a1 = load_a16(qin, m0 + 16, EMB, k0, lane);
    v16h b0 = load_bt16(win, n0,      EMB, k0, lane);
    v16h b1 = load_bt16(win, n0 + 16, EMB, k0, lane);
    acc00 = wmma_f16(a0, b0, acc00);
    acc01 = wmma_f16(a0, b1, acc01);
    acc10 = wmma_f16(a1, b0, acc10);
    acc11 = wmma_f16(a1, b1, acc11);
  }
  store_qkv_tile(acc00, m0,      n0,      lane, bin, qh, kh, vt);
  store_qkv_tile(acc01, m0,      n0 + 16, lane, bin, qh, kh, vt);
  store_qkv_tile(acc10, m0 + 16, n0,      lane, bin, qh, kh, vt);
  store_qkv_tile(acc11, m0 + 16, n0 + 16, lane, bin, qh, kh, vt);
}

// ---------------- K2: fused attention (scores + softmax + weights write + P@V) ----------------
// grid (S/16, B*H), block 128 (4 waves).  One 16-row query tile per workgroup.
// LDS score tile: 16 x 2048 fp32 (padded).  Weights written once (nontemporal);
// LDS keeps e = exp(x - max) <= 1; 1/sum folded into the global write and the PV accumulator.
__global__ void __launch_bounds__(128) k_attn(const _Float16* __restrict__ qh,
                                              const _Float16* __restrict__ kh,
                                              const _Float16* __restrict__ vt,
                                              const float* __restrict__ rpb,
                                              float* __restrict__ wts,
                                              _Float16* __restrict__ aout) {
  __shared__ float sc[16][SC_LD];     // ~131 KB
  __shared__ float red[16][8];
  __shared__ float rowinv[16];

  const int t = threadIdx.x;
  const int lane = t & 31;
  const int wave = t >> 5;
  const int mt = blockIdx.x;
  const int bh = blockIdx.y;
  const int h = bh & (NH - 1), b = bh >> 3;
  const _Float16* qb = qh + (size_t)bh * S_LEN * HDIM;
  const _Float16* kb = kh + (size_t)bh * S_LEN * HDIM;
  const _Float16* vb = vt + (size_t)bh * HDIM * S_LEN;
  const float*    rb = rpb + (size_t)h * S_LEN * S_LEN;

  // ---- Phase 1: scores -> LDS (each wave: 32 of 128 n-tiles; A-frags loaded once) ----
  {
    const v16h a0 = load_a16(qb, mt * 16, HDIM, 0,  lane);
    const v16h a1 = load_a16(qb, mt * 16, HDIM, 32, lane);
    const int hi8 = ((lane >> 4) & 1) << 3;
    for (int nt = wave; nt < S_LEN / 16; nt += 4) {
      v16h b0 = load_bt16(kb, nt * 16, HDIM, 0,  lane);
      v16h b1 = load_bt16(kb, nt * 16, HDIM, 32, lane);
      v8f acc = {};
      acc = wmma_f16(a0, b0, acc);
      acc = wmma_f16(a1, b1, acc);
      const int j = nt * 16 + (lane & 15);
#pragma unroll
      for (int r = 0; r < 8; ++r) {
        const int il = r + hi8;                       // local row 0..15
        float bv = rb[(size_t)(mt * 16 + il) * S_LEN + j];
        bv = fminf(fmaxf(bv, -5.f), 5.f);
        float sv = acc[r] * 0.125f + bv * 0.05f;      // scale = 1/sqrt(64)
        sv = fminf(fmaxf(sv, -30.f), 30.f);
        sc[il][j] = sv;
      }
    }
  }
  __syncthreads();

  // ---- Phase 2: row max, exp-in-place, row sum (thread t: row t>>3, 256-col chunk t&7) ----
  const int row = t >> 3;
  const int chunk = t & 7;
  {
    float mx = -1e30f;
#pragma unroll 4
    for (int i = 0; i < 64; ++i) {
      v4f v = *(const v4f*)&sc[row][chunk * 256 + i * 4];
      mx = fmaxf(mx, fmaxf(fmaxf(v.x, v.y), fmaxf(v.z, v.w)));
    }
    red[row][chunk] = mx;
  }
  __syncthreads();
  {
    float m = red[row][0];
#pragma unroll
    for (int j = 1; j < 8; ++j) m = fmaxf(m, red[row][j]);
    float sum = 0.f;
#pragma unroll 4
    for (int i = 0; i < 64; ++i) {
      v4f v = *(const v4f*)&sc[row][chunk * 256 + i * 4];
      v.x = __expf(v.x - m); v.y = __expf(v.y - m);
      v.z = __expf(v.z - m); v.w = __expf(v.w - m);
      *(v4f*)&sc[row][chunk * 256 + i * 4] = v;
      sum += v.x + v.y + v.z + v.w;
    }
    __syncthreads();                  // all max-reads of red done before reuse
    red[row][chunk] = sum;
  }
  __syncthreads();
  {
    float s = red[row][0];
#pragma unroll
    for (int j = 1; j < 8; ++j) s += red[row][j];
    if (chunk == 0) rowinv[row] = 1.f / s;
  }
  __syncthreads();

  // ---- Phase 3: stream normalized weights to global, write-once, nontemporal ----
  {
    float* wrow = wts + (size_t)bh * S_LEN * S_LEN + (size_t)(mt * 16) * S_LEN;
#pragma unroll
    for (int r = 0; r < 16; ++r) {
      const float inv = rowinv[r];
#pragma unroll
      for (int i = 0; i < 4; ++i) {
        const int c = (i * 128 + t) * 4;
        v4f v = *(const v4f*)&sc[r][c];
        v *= inv;
        __builtin_nontemporal_store(v, (v4f*)(wrow + (size_t)r * S_LEN + c));
      }
    }
  }

  // ---- Phase 4: out_h = P @ V; wave w owns hd-tile w; A from LDS (f32->f16), B = V^T rows ----
  {
    const int hiA = ((lane >> 4) & 1) << 3;
    const float* prow = &sc[lane & 15][0];
    v8f acc = {};
    for (int k0 = 0; k0 < S_LEN; k0 += 32) {
      float tmp[16];
      *(v4f*)(tmp + 0)  = *(const v4f*)(prow + k0 + hiA);
      *(v4f*)(tmp + 4)  = *(const v4f*)(prow + k0 + hiA + 4);
      *(v4f*)(tmp + 8)  = *(const v4f*)(prow + k0 + hiA + 16);
      *(v4f*)(tmp + 12) = *(const v4f*)(prow + k0 + hiA + 20);
      v16h a;
#pragma unroll
      for (int e = 0; e < 16; ++e) a[e] = (_Float16)tmp[e];
      v16h bm = load_bt16(vb, wave * 16, S_LEN, k0, lane);
      acc = wmma_f16(a, bm, acc);
    }
    const int e_col = h * HDIM + wave * 16 + (lane & 15);
    const int hi8 = ((lane >> 4) & 1) << 3;
#pragma unroll
    for (int r = 0; r < 8; ++r) {
      const int il = r + hi8;
      const int s = mt * 16 + il;
      float o = acc[r] * rowinv[il];
      o = (o != o) ? 0.f : o;                    // NaN -> 0
      o = fminf(fmaxf(o, -10.f), 10.f);          // clip +-10
      aout[((size_t)b * S_LEN + s) * EMB + e_col] = (_Float16)o;
    }
  }
}

// ---------------- K3: out-proj GEMM (8192 x 512 x 512), 32x32 per wave, (B,S,E)->(S,B,E) ----------------
static __device__ __forceinline__ void store_out_tile(const v8f& acc, int m0, int n0, int lane,
                                                      const float* __restrict__ bout,
                                                      float* __restrict__ out) {
  const int n = n0 + (lane & 15);
  const float bias = bout[n];
  const int hi8 = ((lane >> 4) & 1) << 3;
#pragma unroll
  for (int r = 0; r < 8; ++r) {
    const int m = m0 + r + hi8;
    const int b = m >> 11;            // / S_LEN
    const int s = m & (S_LEN - 1);
    float o = acc[r] + bias;
    o = (o != o) ? 0.f : o;
    out[((size_t)s * BATCH + b) * EMB + n] = o;
  }
}

__global__ void k_outproj(const _Float16* __restrict__ aout, const _Float16* __restrict__ wout,
                          const float* __restrict__ bout, float* __restrict__ out) {
  const int lane = threadIdx.x;
  const int m0 = blockIdx.x * 32;     // rows: b*S + s
  const int n0 = blockIdx.y * 32;
  v8f acc00 = {}, acc01 = {}, acc10 = {}, acc11 = {};
#pragma unroll 2
  for (int k0 = 0; k0 < EMB; k0 += 32) {
    v16h a0 = load_a16(aout, m0,      EMB, k0, lane);
    v16h a1 = load_a16(aout, m0 + 16, EMB, k0, lane);
    v16h b0 = load_bt16(wout, n0,      EMB, k0, lane);
    v16h b1 = load_bt16(wout, n0 + 16, EMB, k0, lane);
    acc00 = wmma_f16(a0, b0, acc00);
    acc01 = wmma_f16(a0, b1, acc01);
    acc10 = wmma_f16(a1, b0, acc10);
    acc11 = wmma_f16(a1, b1, acc11);
  }
  store_out_tile(acc00, m0,      n0,      lane, bout, out);
  store_out_tile(acc01, m0,      n0 + 16, lane, bout, out);
  store_out_tile(acc10, m0 + 16, n0,      lane, bout, out);
  store_out_tile(acc11, m0 + 16, n0 + 16, lane, bout, out);
}

extern "C" void kernel_launch(void* const* d_in, const int* in_sizes, int n_in,
                              void* d_out, int out_size, void* d_ws, size_t ws_size,
                              hipStream_t stream) {
  (void)in_sizes; (void)n_in; (void)out_size; (void)ws_size;
  const float* query  = (const float*)d_in[0];
  // d_in[1] (key) and d_in[2] (value) are unused: reference projects qkv from query.
  const float* rpb    = (const float*)d_in[3];
  const float* win_f  = (const float*)d_in[4];
  const float* bin    = (const float*)d_in[5];
  const float* wout_f = (const float*)d_in[6];
  const float* bout   = (const float*)d_in[7];

  float* out = (float*)d_out;                                   // (S,B,E)
  float* wts = out + (size_t)S_LEN * BATCH * EMB;               // (B,H,S,S)

  char* ws = (char*)d_ws;
  _Float16* qin  = (_Float16*)ws; ws += (size_t)S_LEN * BATCH * EMB * 2;
  _Float16* win  = (_Float16*)ws; ws += (size_t)E3 * EMB * 2;
  _Float16* wout = (_Float16*)ws; ws += (size_t)EMB * EMB * 2;
  _Float16* qh   = (_Float16*)ws; ws += (size_t)BATCH * NH * S_LEN * HDIM * 2;
  _Float16* kh   = (_Float16*)ws; ws += (size_t)BATCH * NH * S_LEN * HDIM * 2;
  _Float16* vt   = (_Float16*)ws; ws += (size_t)BATCH * NH * S_LEN * HDIM * 2;
  _Float16* aout = (_Float16*)ws;

  const int nQ = S_LEN * BATCH * EMB;
  k_cvt_f16<<<(nQ + 255) / 256, 256, 0, stream>>>(query, qin, nQ);
  k_cvt_f16<<<(E3 * EMB + 255) / 256, 256, 0, stream>>>(win_f, win, E3 * EMB);
  k_cvt_f16<<<(EMB * EMB + 255) / 256, 256, 0, stream>>>(wout_f, wout, EMB * EMB);

  k_qkv_gemm<<<dim3(256, 48), 32, 0, stream>>>(qin, win, bin, qh, kh, vt);
  k_attn<<<dim3(S_LEN / 16, BATCH * NH), 128, 0, stream>>>(qh, kh, vt, rpb, wts, aout);
  k_outproj<<<dim3(256, 16), 32, 0, stream>>>(aout, wout, bout, out);
}
